// DeformableConv2dWrapper_78323023610235
// MI455X (gfx1250) — matrix operand order repaired and checked by reading, not compiled, not run on hardware
//
#include <hip/hip_runtime.h>
#include <hip/hip_bf16.h>
#include <math.h>

// ---------------------------------------------------------------------------
// Deformable Conv2d for MI455X (gfx1250, wave32, WMMA).
// B=4, C=64, H=W=128, O=64, K=3, S=1, P=1.
//   k1: repack w_off -> f16 A (32x576, zero-padded, K = k*64+c)
//   k2: repack w_dcn -> f16 A (64x576, K = k2*64+c)
//   k3: offset conv = im2col->LDS f16 panel + WMMA f32_16x16x32_f16
//   k4: bilinear sample -> LDS f16 panel + WMMA f32_16x16x32_f16 GEMM
// All matrix math on the XDL/WMMA pipes; fp32 accumulation throughout.
// ---------------------------------------------------------------------------

typedef __attribute__((ext_vector_type(16))) _Float16 v16h;
typedef __attribute__((ext_vector_type(8)))  _Float16 v8h;
typedef __attribute__((ext_vector_type(2)))  _Float16 h2;
typedef __attribute__((ext_vector_type(8)))  float    v8f;

#define BATCH 4
#define CIN   64
#define COUT  64
#define Hdim  128
#define Wdim  128
#define HW    (Hdim * Wdim)
#define K2    9
#define OFFC  18            // 2 * K * K
#define OFFM  32            // OFFC padded to WMMA M granularity
#define KDIM  (CIN * K2)    // 576 (GEMM K)
#define BSTRIDE 584         // LDS row stride in halves (16B-aligned rows)

// Shared helper: build one 16x16 WMMA step fragment pair and accumulate.
// A row base = per-lane M row (global), B row base = per-lane pixel row in LDS.
// ISA 16-bit A layout: elem h -> K = (h/8)*16 + lhalf*8 + h%8
// ISA 16-bit B layout: elem h -> K = (h/8)*16 + lhalf*8 + h%8 (same as A)
__device__ __forceinline__ v8f wmma_step(const _Float16* __restrict__ Arow,
                                         const _Float16* __restrict__ Brow,
                                         int kb, int lhalf, v8f acc) {
    const int a0 = kb + lhalf * 8;
    v8h lo = *(const v8h*)(Arow + a0);
    v8h hi = *(const v8h*)(Arow + a0 + 16);
    v16h a = __builtin_shufflevector(lo, hi, 0,1,2,3,4,5,6,7,
                                             8,9,10,11,12,13,14,15);
    const _Float16* bp = Brow + kb + lhalf * 8;
    v8h blo = *(const v8h*)(bp);
    v8h bhi = *(const v8h*)(bp + 16);
    v16h bf = __builtin_shufflevector(blo, bhi, 0,1,2,3,4,5,6,7,
                                                8,9,10,11,12,13,14,15);
    return __builtin_amdgcn_wmma_f32_16x16x32_f16(
               false, a, false, bf, (short)0, acc, false, false);
}

// ---------------------------------------------------------------------------
// Kernel 1: repack w_off (18,64,3,3) fp32 -> f16 32x576 rows, K = k*64 + c,
// rows 18..31 zero.
// ---------------------------------------------------------------------------
__global__ __launch_bounds__(256)
void woff_repack_kernel(const float* __restrict__ w_off, _Float16* __restrict__ wOffA) {
    const int idx = blockIdx.x * 256 + threadIdx.x;
    if (idx >= OFFM * KDIM) return;
    const int o  = idx / KDIM;
    const int kk = idx - o * KDIM;
    const int k  = kk >> 6;         // kernel tap 0..8
    const int c  = kk & 63;         // input channel
    wOffA[idx] = (o < OFFC) ? (_Float16)w_off[o * KDIM + c * K2 + k] : (_Float16)0.f;
}

// ---------------------------------------------------------------------------
// Kernel 2: repack w_dcn (O,C,3,3) fp32 -> f16 rows indexed by K = k2*64 + c
// ---------------------------------------------------------------------------
__global__ __launch_bounds__(256)
void wconv_kernel(const float* __restrict__ w_dcn, _Float16* __restrict__ wA) {
    const int idx = blockIdx.x * 256 + threadIdx.x;
    if (idx >= COUT * KDIM) return;
    const int o  = idx / KDIM;
    const int kk = idx - o * KDIM;
    const int k2 = kk >> 6;
    const int c  = kk & 63;
    wA[idx] = (_Float16)w_dcn[o * KDIM + c * K2 + k2];
}

// ---------------------------------------------------------------------------
// Kernel 3: offset conv as implicit GEMM on WMMA.
// Block: 256 threads (8 wave32). Tile: 32 pixels x 18(->32) offset channels.
// Phase 1 (all waves): im2col panel sB[pixel][k*64+c] as f16 (channel-paired).
// Phase 2 (waves 0..3): 16x16 WMMA tiles, fp32 accumulate, + bias, store fp32.
// ---------------------------------------------------------------------------
__global__ __launch_bounds__(256)
void offset_wmma_kernel(const float* __restrict__ x,
                        const _Float16* __restrict__ wOffA,
                        const float* __restrict__ b_off,
                        float* __restrict__ offset) {
    __shared__ _Float16 sB[32 * BSTRIDE];      // 36.5 KB

    const int tile = blockIdx.x;               // b*512 + y*4 + nt
    const int nt = tile & 3;
    const int y  = (tile >> 2) & (Hdim - 1);
    const int b  = tile >> 9;
    const int x0 = nt * 32;
    const int t  = threadIdx.x;

    // ---- Phase 1: im2col fill (32 pixels x 32 channel-pairs = 1024 jobs) ----
    const float* xb = x + (size_t)b * CIN * HW;
    for (int pair = t; pair < 32 * (CIN / 2); pair += 256) {
        const int pix = pair & 31;
        const int c   = (pair >> 5) * 2;       // even channel of the pair
        const int xx  = x0 + pix;
        const float* xc0 = xb + (size_t)c * HW;
        const float* xc1 = xc0 + HW;
        const int rowBase = pix * BSTRIDE + c;
        #pragma unroll
        for (int kh = 0; kh < 3; ++kh) {
            const int  yy  = y - 1 + kh;
            const bool yok = (unsigned)yy < (unsigned)Hdim;
            #pragma unroll
            for (int kw = 0; kw < 3; ++kw) {
                const int  xk = xx - 1 + kw;
                const bool ok = yok && (unsigned)xk < (unsigned)Wdim;
                const int  gi = yy * Wdim + xk;
                const float v0 = ok ? xc0[gi] : 0.0f;
                const float v1 = ok ? xc1[gi] : 0.0f;
                *reinterpret_cast<h2*>(&sB[rowBase + (kh * 3 + kw) * 64]) =
                    (h2){(_Float16)v0, (_Float16)v1};
            }
        }
    }
    __syncthreads();

    // ---- Phase 2: WMMA (waves 0..3; wave-granular branch, EXEC all-1s) ----
    if (t < 128) {
        const int wv    = t >> 5;       // wave 0..3
        const int lane  = t & 31;
        const int mT    = wv & 1;       // M tile: rows mT*16
        const int nT    = wv >> 1;      // N tile: pixels nT*16
        const int lhalf = lane >> 4;
        const int lm    = lane & 15;

        const _Float16* Arow = wOffA + (size_t)(mT * 16 + lm) * KDIM;
        const _Float16* Brow = &sB[(nT * 16 + lm) * BSTRIDE];

        v8f acc = {};
        #pragma unroll
        for (int ks = 0; ks < KDIM / 32; ++ks)
            acc = wmma_step(Arow, Brow, ks * 32, lhalf, acc);

        const int xx = x0 + nT * 16 + lm;
        #pragma unroll
        for (int r = 0; r < 8; ++r) {
            const int oc = mT * 16 + r + lhalf * 8;
            if (oc < OFFC)
                offset[(((size_t)b * OFFC + oc) * Hdim + y) * Wdim + xx] =
                    acc[r] + b_off[oc];
        }
    }
}

// ---------------------------------------------------------------------------
// Kernel 4: bilinear sample into LDS f16 panel, then WMMA GEMM.
// Block: 256 threads (8 wave32). Tile: 32 pixels x 64 output channels.
// ---------------------------------------------------------------------------
__global__ __launch_bounds__(256)
void deform_wmma_kernel(const float* __restrict__ x,
                        const float* __restrict__ offset,
                        const _Float16* __restrict__ wA,
                        const float* __restrict__ b_dcn,
                        float* __restrict__ out) {
    __shared__ _Float16 sB[32 * BSTRIDE];      // K x N panel, [pixel][K], 36.5 KB

    const int tile = blockIdx.x;               // b*512 + y*4 + nt
    const int nt = tile & 3;
    const int y  = (tile >> 2) & (Hdim - 1);
    const int b  = tile >> 9;
    const int x0 = nt * 32;
    const int t  = threadIdx.x;

    // ---------------- Phase 1: bilinear sampling -> sB ----------------
    const float* xb = x + (size_t)b * CIN * HW;
    for (int pair = t; pair < 32 * K2; pair += 256) {
        const int pix = pair & 31;
        const int k2  = pair >> 5;
        const int xx  = x0 + pix;

        const size_t obase = (((size_t)b * OFFC + 2 * k2) * Hdim + y) * Wdim + xx;
        const float oy = offset[obase];
        const float ox = offset[obase + HW];   // channel 2*k2+1

        const float py = (float)(y - 1 + k2 / 3) + oy;
        const float px = (float)(xx - 1 + k2 % 3) + ox;
        const float y0f = floorf(py), x0f = floorf(px);
        const float dy = py - y0f,    dx = px - x0f;
        const int iy0 = (int)y0f, ix0 = (int)x0f;
        const int iy1 = iy0 + 1,  ix1 = ix0 + 1;

        const float vy0 = ((unsigned)iy0 < (unsigned)Hdim) ? 1.0f : 0.0f;
        const float vy1 = ((unsigned)iy1 < (unsigned)Hdim) ? 1.0f : 0.0f;
        const float vx0 = ((unsigned)ix0 < (unsigned)Wdim) ? 1.0f : 0.0f;
        const float vx1 = ((unsigned)ix1 < (unsigned)Wdim) ? 1.0f : 0.0f;
        const int cy0 = min(max(iy0, 0), Hdim - 1);
        const int cy1 = min(max(iy1, 0), Hdim - 1);
        const int cx0 = min(max(ix0, 0), Wdim - 1);
        const int cx1 = min(max(ix1, 0), Wdim - 1);

        // fold per-corner validity into the bilinear weights
        const float w00 = (1.f - dy) * (1.f - dx) * vy0 * vx0;
        const float w01 = (1.f - dy) * dx         * vy0 * vx1;
        const float w10 = dy         * (1.f - dx) * vy1 * vx0;
        const float w11 = dy         * dx         * vy1 * vx1;
        const int i00 = cy0 * Wdim + cx0, i01 = cy0 * Wdim + cx1;
        const int i10 = cy1 * Wdim + cx0, i11 = cy1 * Wdim + cx1;

        const int rowBase = pix * BSTRIDE + k2 * 64;   // sB[pixel][k2*64 + c]
        for (int c = 0; c < CIN; c += 2) {
            const float* p0 = xb + (size_t)c * HW;
            const float* p1 = p0 + HW;
            const float s0 = p0[i00]*w00 + p0[i01]*w01 + p0[i10]*w10 + p0[i11]*w11;
            const float s1 = p1[i00]*w00 + p1[i01]*w01 + p1[i10]*w10 + p1[i11]*w11;
            *reinterpret_cast<h2*>(&sB[rowBase + c]) = (h2){(_Float16)s0, (_Float16)s1};
        }
    }
    __syncthreads();

    // ---------------- Phase 2: WMMA GEMM (uniform flow, EXEC all-1) ----------
    const int wv    = t >> 5;       // wave 0..7
    const int lane  = t & 31;
    const int mT    = wv & 3;       // M tile: output channels mT*16..+15
    const int nT    = wv >> 2;      // N tile: pixels nT*16..+15
    const int lhalf = lane >> 4;
    const int lm    = lane & 15;

    const _Float16* Arow = wA + (size_t)(mT * 16 + lm) * KDIM; // per-lane M row
    const _Float16* Brow = &sB[(nT * 16 + lm) * BSTRIDE];      // per-lane pixel

    v8f acc = {};
    #pragma unroll
    for (int ks = 0; ks < KDIM / 32; ++ks)         // 18 WMMA K-steps
        acc = wmma_step(Arow, Brow, ks * 32, lhalf, acc);

    // D layout: reg r -> M = mT*16 + r + lhalf*8, N = nT*16 + lm
    const int xx = x0 + nT * 16 + lm;
    #pragma unroll
    for (int r = 0; r < 8; ++r) {
        const int o = mT * 16 + r + lhalf * 8;
        out[(((size_t)b * COUT + o) * Hdim + y) * Wdim + xx] = acc[r] + b_dcn[o];
    }
}

// ---------------------------------------------------------------------------
extern "C" void kernel_launch(void* const* d_in, const int* in_sizes, int n_in,
                              void* d_out, int out_size, void* d_ws, size_t ws_size,
                              hipStream_t stream) {
    const float* x     = (const float*)d_in[0];
    const float* w_off = (const float*)d_in[1];
    const float* b_off = (const float*)d_in[2];
    const float* w_dcn = (const float*)d_in[3];
    const float* b_dcn = (const float*)d_in[4];
    float* out = (float*)d_out;

    // workspace layout
    char* ws = (char*)d_ws;
    float*    offs  = (float*)ws;                                   // 4.72 MB
    ws += (size_t)BATCH * OFFC * HW * sizeof(float);
    _Float16* wA    = (_Float16*)ws;                                // 73.7 KB
    ws += (size_t)COUT * KDIM * sizeof(_Float16);
    _Float16* wOffA = (_Float16*)ws;                                // 36.9 KB

    // 1) repack w_off -> padded f16 A matrix
    woff_repack_kernel<<<(OFFM * KDIM + 255) / 256, 256, 0, stream>>>(w_off, wOffA);

    // 2) repack w_dcn -> f16 A matrix
    wconv_kernel<<<(COUT * KDIM + 255) / 256, 256, 0, stream>>>(w_dcn, wA);

    // 3) offset conv on WMMA: B*H*(W/32) = 2048 blocks
    offset_wmma_kernel<<<BATCH * Hdim * (Wdim / 32), 256, 0, stream>>>(
        x, wOffA, b_off, offs);

    // 4) deformable sample + WMMA GEMM: 2048 blocks
    deform_wmma_kernel<<<BATCH * Hdim * (Wdim / 32), 256, 0, stream>>>(
        x, offs, wA, b_dcn, out);
}
